// DenoiseFlow_53523882443631
// MI455X (gfx1250) — matrix operand order repaired and checked by reading, not compiled, not run on hardware
//
#include <hip/hip_runtime.h>
#include <math.h>

// ---------------------------------------------------------------------------
// CDNA5 (gfx1250) implementation of the DenoiseFlow reference.
// Dense layers: f32 -> WMMA-ready f16 operand packing kernels, then a clean
// v_wmma_f32_16x16x32_f16 GEMM whose inner loop is two b128-pair loads + WMMA.
// Working set ~25MB -> L2 resident on MI455X (192MB L2).
// ---------------------------------------------------------------------------

typedef __attribute__((ext_vector_type(16))) _Float16 v16h;
typedef __attribute__((ext_vector_type(8)))  float    v8f;

#define IN_CHn   3
#define AUG_CHn  20
#define AUGHn    10          // AUG_CH/2
#define IDIMn    23
#define C1n      12
#define C2n      11
#define HIDn     64
#define Bn       4
#define Nptsn    1024
#define Pn       4096        // B*N points
#define NFLOWn   12
#define CLAMPc   1.9f
#define PIc      3.14159265358979323846f
#define LOG2PIc  1.8378770664093453f
#define KMAXn    24
#define CHUNKP   1024        // points per KNN-GEMM chunk (rows = CHUNKP*k)

__device__ __forceinline__ float softclamp_d(float s) {
    return (2.0f * CLAMPc / PIc) * atanf(s / CLAMPc);
}

// ---------------------------------------------------------------------------
// Operand packing into WMMA-native layouts (ISA 7.12.2), zero-padded, f16.
// A (16x32 f16): lane L holds row M=L&15, kbase=(L<16)?0:8;
//   half e: v=e>>1 ; K = ((v&4)?16:0) + kbase + 2*(v&3) + (e&1)
// B (32x16 f16): lane L holds col N=L&15, kbase=(L<16)?0:8 ; half e: K=kbase+e+((e<8)?0:8)
// Packed as Ap[(mt*Kt+kt)*32 + lane][16], Bp[(nt*Kt+kt)*32 + lane][16].
// Branchless padding: clamp the index, select 0 with v_cndmask.
// ---------------------------------------------------------------------------
__global__ void pack_a_kernel(const float* __restrict__ X, int M, int K, int Kt,
                              _Float16* __restrict__ Ap) {
    int t = blockIdx.x * blockDim.x + threadIdx.x;
    int total = (M >> 4) * Kt * 32;
    if (t >= total) return;
    int lane = t & 31;
    int tile = t >> 5;
    int kt = tile % Kt, mt = tile / Kt;
    int r  = (mt << 4) + (lane & 15);          // M is always a multiple of 16
    int kb = (lane < 16) ? 0 : 8;
    const float* xr = X + (size_t)r * K;
    v16h a;
#pragma unroll
    for (int e = 0; e < 16; ++e) {
        int v = e >> 1;
        int k = (kt << 5) + ((v & 4) ? 16 : 0) + kb + 2 * (v & 3) + (e & 1);
        int kc = (k < K) ? k : (K - 1);        // clamped unconditional load
        float val = xr[kc];
        a[e] = (_Float16)((k < K) ? val : 0.0f);
    }
    ((v16h*)Ap)[t] = a;                        // 32B vector store
}

__global__ void pack_b_kernel(const float* __restrict__ W, int K, int N, int Kt,
                              _Float16* __restrict__ Bp) {
    int t = blockIdx.x * blockDim.x + threadIdx.x;
    int ntiles = (N + 15) >> 4;
    int total = ntiles * Kt * 32;
    if (t >= total) return;
    int lane = t & 31;
    int tile = t >> 5;
    int kt = tile % Kt, nt = tile / Kt;
    int c  = (nt << 4) + (lane & 15);
    int cc = (c < N) ? c : (N - 1);
    int kb = (kt << 5) + ((lane < 16) ? 0 : 8);
    v16h b;
#pragma unroll
    for (int e = 0; e < 16; ++e) {
        int k  = kb + ((e < 8) ? e : (e + 8));
        int kc = (k < K) ? k : (K - 1);
        float val = W[(size_t)kc * N + cc];
        b[e] = (_Float16)((k < K && c < N) ? val : 0.0f);
    }
    ((v16h*)Bp)[t] = b;
}

// ---------------------------------------------------------------------------
// WMMA GEMM on packed operands: Y[M,N] = act(A @ B + bias).
// One wave per 16x16 tile; inner loop = 2 vector loads + 1 v_wmma.
// C/D layout: vgpr i -> row = i + ((lane>>4)<<3), col = lane&15.
// ---------------------------------------------------------------------------
__global__ void gemm_wmma_kernel(const _Float16* __restrict__ Ap,
                                 const _Float16* __restrict__ Bp,
                                 const float* __restrict__ bias,
                                 float* __restrict__ Y,
                                 int M, int N, int Kt, int relu) {
    int wave = (int)((blockIdx.x * blockDim.x + threadIdx.x) >> 5);
    int lane = threadIdx.x & 31;
    int mtiles = M >> 4;                       // M multiple of 16 by construction
    int ntiles = (N + 15) >> 4;
    if (wave >= mtiles * ntiles) return;       // wave-uniform exit (EXEC full)
    int mt = wave / ntiles, nt = wave % ntiles;

    const v16h* Abase = (const v16h*)Ap + (size_t)mt * Kt * 32 + lane;
    const v16h* Bbase = (const v16h*)Bp + (size_t)nt * Kt * 32 + lane;

    v8f acc = {};
    for (int kt = 0; kt < Kt; ++kt) {
        v16h a = Abase[(size_t)kt * 32];
        v16h b = Bbase[(size_t)kt * 32];
        acc = __builtin_amdgcn_wmma_f32_16x16x32_f16(
            /*neg_a=*/false, a, /*neg_b=*/false, b,
            /*c_mod=*/(short)0, acc, /*reuse_a=*/false, /*reuse_b=*/false);
    }

    int c  = (nt << 4) + (lane & 15);
    float bv = (bias != nullptr && c < N) ? bias[c] : 0.0f;
    int rb = (mt << 4) + ((lane >> 4) << 3);
#pragma unroll
    for (int i = 0; i < 8; ++i) {
        if (c < N) {
            float v = acc[i] + bv;
            if (relu) v = fmaxf(v, 0.0f);
            Y[(size_t)(rb + i) * N + c] = v;
        }
    }
}

// ---------------------------------------------------------------------------
// Brute-force KNN (top-k smallest squared distance, self included).
// ---------------------------------------------------------------------------
__global__ void knn_kernel(const float* __restrict__ F, int C, int k,
                           int* __restrict__ idx) {
    int p = blockIdx.x * blockDim.x + threadIdx.x;
    if (p >= Pn) return;
    int b = p / Nptsn, n = p % Nptsn;
    const float* fb = F + (size_t)b * Nptsn * C;
    float fi[IDIMn];
    for (int c = 0; c < C; ++c) fi[c] = fb[(size_t)n * C + c];
    float bd[KMAXn]; int bi[KMAXn];
    for (int j = 0; j < k; ++j) { bd[j] = 3.4e38f; bi[j] = 0; }
    for (int m = 0; m < Nptsn; ++m) {
        const float* fm = fb + (size_t)m * C;
        float d = 0.0f;
        for (int c = 0; c < C; ++c) { float df = fi[c] - fm[c]; d += df * df; }
        if (d < bd[k - 1]) {
            int j = k - 1;
            while (j > 0 && bd[j - 1] > d) { bd[j] = bd[j - 1]; bi[j] = bi[j - 1]; --j; }
            bd[j] = d; bi[j] = m;
        }
    }
    for (int j = 0; j < k; ++j) idx[(size_t)p * k + j] = bi[j];
}

// Build KNN-MLP input rows: [center(12) | neighbor(12) | neighbor-center(12)]
__global__ void knn_build_kernel(const float* __restrict__ X1,
                                 const int* __restrict__ idx, int k,
                                 int p0, int np, float* __restrict__ dst) {
    int r = blockIdx.x * blockDim.x + threadIdx.x;
    if (r >= np * k) return;
    int p = p0 + r / k, kk = r % k;
    int b = p / Nptsn;
    int nb = idx[(size_t)p * k + kk];
    const float* fc = X1 + (size_t)p * C1n;
    const float* fn = X1 + ((size_t)b * Nptsn + nb) * C1n;
    float* d = dst + (size_t)r * (3 * C1n);
    for (int c = 0; c < C1n; ++c) {
        float a = fc[c], v = fn[c];
        d[c] = a; d[C1n + c] = v; d[2 * C1n + c] = v - a;
    }
}

__global__ void rowmax_kernel(const float* __restrict__ src, int k, int C,
                              int p0, int np, float* __restrict__ dst) {
    int t = blockIdx.x * blockDim.x + threadIdx.x;
    if (t >= np * C) return;
    int pl = t / C, c = t % C;
    float m = -3.4e38f;
    for (int j = 0; j < k; ++j)
        m = fmaxf(m, src[((size_t)pl * k + j) * C + c]);
    dst[(size_t)(p0 + pl) * C + c] = m;
}

// ---------------------------------------------------------------------------
// Small elementwise / slicing kernels
// ---------------------------------------------------------------------------
__global__ void slice12_kernel(const float* __restrict__ src, float* __restrict__ dst) {
    int t = blockIdx.x * blockDim.x + threadIdx.x;
    if (t >= Pn * C1n) return;
    int p = t / C1n, c = t % C1n;
    dst[t] = src[(size_t)p * IDIMn + c];
}

__global__ void concat_xy_kernel(const float* __restrict__ x,
                                 const float* __restrict__ y,
                                 float* __restrict__ dst) {
    int t = blockIdx.x * blockDim.x + threadIdx.x;
    if (t >= Pn * IDIMn) return;
    int p = t / IDIMn, c = t % IDIMn;
    dst[t] = (c < IN_CHn) ? x[(size_t)p * IN_CHn + c]
                          : y[(size_t)p * AUG_CHn + (c - IN_CHn)];
}

__global__ void reverse_kernel(const float* __restrict__ src, float* __restrict__ dst) {
    int t = blockIdx.x * blockDim.x + threadIdx.x;
    if (t >= Pn * IDIMn) return;
    int p = t / IDIMn, c = t % IDIMn;
    dst[(size_t)p * IDIMn + c] = src[(size_t)p * IDIMn + (IDIMn - 1 - c)];
}

__global__ void affine_fwd_kernel(float* __restrict__ H,
                                  const float* __restrict__ logs,
                                  const float* __restrict__ bias) {
    int t = blockIdx.x * blockDim.x + threadIdx.x;
    if (t >= Pn * IDIMn) return;
    int c = t % IDIMn;
    H[t] = H[t] * expf(logs[c]) + bias[c];
}

__global__ void affine_inv_kernel(float* __restrict__ H,
                                  const float* __restrict__ logs,
                                  const float* __restrict__ bias) {
    int t = blockIdx.x * blockDim.x + threadIdx.x;
    if (t >= Pn * IDIMn) return;
    int c = t % IDIMn;
    H[t] = (H[t] - bias[c]) * expf(-logs[c]);
}

__global__ void zero_tail_kernel(float* __restrict__ H) {
    int t = blockIdx.x * blockDim.x + threadIdx.x;
    if (t >= Pn * IN_CHn) return;
    int p = t / IN_CHn, c = t % IN_CHn;
    H[(size_t)p * IDIMn + (IDIMn - IN_CHn) + c] = 0.0f;
}

__global__ void out_kernel(const float* __restrict__ z, float* __restrict__ out) {
    int t = blockIdx.x * blockDim.x + threadIdx.x;
    if (t >= Pn * IN_CHn) return;
    int p = t / IN_CHn, c = t % IN_CHn;
    out[t] = z[(size_t)p * IDIMn + c];
}

// ---------------------------------------------------------------------------
// Coupling combine (fwd accumulates per-batch sum(s) via LDS + global atomics)
// ---------------------------------------------------------------------------
__global__ void couple_fwd_kernel(float* __restrict__ H,
                                  const float* __restrict__ S,
                                  const float* __restrict__ T,
                                  float* __restrict__ ldj) {
    __shared__ float bl[Bn];
    if (threadIdx.x < Bn) bl[threadIdx.x] = 0.0f;
    __syncthreads();
    int t = blockIdx.x * blockDim.x + threadIdx.x;
    float sacc = 0.0f; int bidx = -1;
    if (t < Pn * C2n) {
        int p = t / C2n, c = t % C2n;
        bidx = p / Nptsn;
        float s = softclamp_d(S[t]);
        float* hp = &H[(size_t)p * IDIMn + C1n + c];
        *hp = (*hp) * expf(s) + T[t];
        sacc = s;
    }
    if (bidx >= 0) atomicAdd(&bl[bidx], sacc);
    __syncthreads();
    if (threadIdx.x < Bn) atomicAdd(&ldj[threadIdx.x], bl[threadIdx.x]);
}

__global__ void couple_inv_kernel(float* __restrict__ H,
                                  const float* __restrict__ S,
                                  const float* __restrict__ T) {
    int t = blockIdx.x * blockDim.x + threadIdx.x;
    if (t >= Pn * C2n) return;
    int p = t / C2n, c = t % C2n;
    float s = softclamp_d(S[t]);
    float* hp = &H[(size_t)p * IDIMn + C1n + c];
    *hp = ((*hp) - T[t]) * expf(-s);
}

// ---------------------------------------------------------------------------
// Augmentation steps
// ---------------------------------------------------------------------------
__global__ void aug_concat_kernel(const float* __restrict__ y,
                                  const float* __restrict__ feat,
                                  float* __restrict__ dst, int rev) {
    const int CC = AUGHn + HIDn;  // 74
    int t = blockIdx.x * blockDim.x + threadIdx.x;
    if (t >= Pn * CC) return;
    int p = t / CC, c = t % CC;
    dst[t] = (c < AUGHn) ? y[(size_t)p * AUG_CHn + (rev ? AUGHn : 0) + c]
                         : feat[(size_t)p * HIDn + (c - AUGHn)];
}

__global__ void aug_combine_kernel(float* __restrict__ y,
                                   const float* __restrict__ S,
                                   const float* __restrict__ T,
                                   int rev, float* __restrict__ stepldj) {
    __shared__ float bl[Bn];
    if (threadIdx.x < Bn) bl[threadIdx.x] = 0.0f;
    __syncthreads();
    int t = blockIdx.x * blockDim.x + threadIdx.x;
    float sacc = 0.0f; int bidx = -1;
    if (t < Pn * AUGHn) {
        int p = t / AUGHn, c = t % AUGHn;
        bidx = p / Nptsn;
        float s = softclamp_d(S[t]);
        float* tr = &y[(size_t)p * AUG_CHn + (rev ? 0 : AUGHn) + c];
        *tr = (*tr) * expf(s) + T[t];
        sacc = s;
    }
    if (bidx >= 0) atomicAdd(&bl[bidx], sacc);
    __syncthreads();
    if (threadIdx.x < Bn) atomicAdd(&stepldj[threadIdx.x], bl[threadIdx.x]);
}

// ---------------------------------------------------------------------------
// BatchNorm over (B,N) per channel: LDS tree reductions.
// ---------------------------------------------------------------------------
__global__ void bn_stats_kernel(const float* __restrict__ X, int C,
                                float* __restrict__ mean, float* __restrict__ var) {
    int c = blockIdx.x;
    __shared__ float ssum[256], ssq[256];
    float s = 0.0f, q = 0.0f;
    for (int r = threadIdx.x; r < Pn; r += blockDim.x) {
        float v = X[(size_t)r * C + c]; s += v; q += v * v;
    }
    ssum[threadIdx.x] = s; ssq[threadIdx.x] = q;
    __syncthreads();
    for (int off = 128; off > 0; off >>= 1) {
        if ((int)threadIdx.x < off) {
            ssum[threadIdx.x] += ssum[threadIdx.x + off];
            ssq[threadIdx.x]  += ssq[threadIdx.x + off];
        }
        __syncthreads();
    }
    if (threadIdx.x == 0) {
        float m = ssum[0] / (float)Pn;
        mean[c] = m;
        var[c]  = ssq[0] / (float)Pn - m * m;
    }
}

__global__ void bn_apply_relu_kernel(float* __restrict__ X, int C,
                                     const float* __restrict__ mean,
                                     const float* __restrict__ var,
                                     const float* __restrict__ g,
                                     const float* __restrict__ be) {
    int t = blockIdx.x * blockDim.x + threadIdx.x;
    if (t >= Pn * C) return;
    int c = t % C;
    float v = (X[t] - mean[c]) * rsqrtf(var[c] + 1e-5f) * g[c] + be[c];
    X[t] = fmaxf(v, 0.0f);
}

// ---------------------------------------------------------------------------
// Gauss-Jordan inversion + log|det| for the 12 mixing matrices (LDS, 1 blk ea)
// ---------------------------------------------------------------------------
__global__ void invert_kernel(const float* __restrict__ Wpack,
                              float* __restrict__ invW,
                              float* __restrict__ logdet) {
    int f = blockIdx.x;
    __shared__ float A[IDIMn][2 * IDIMn];
    __shared__ float ld;
    int tid = threadIdx.x;
    for (int e = tid; e < IDIMn * 2 * IDIMn; e += blockDim.x) {
        int r = e / (2 * IDIMn), c = e % (2 * IDIMn);
        A[r][c] = (c < IDIMn) ? Wpack[(size_t)f * IDIMn * IDIMn + r * IDIMn + c]
                              : ((c - IDIMn == r) ? 1.0f : 0.0f);
    }
    if (tid == 0) ld = 0.0f;
    __syncthreads();
    for (int i = 0; i < IDIMn; ++i) {
        if (tid == 0) ld += logf(fabsf(A[i][i]));
        __syncthreads();
        float piv = A[i][i];
        for (int c = tid; c < 2 * IDIMn; c += blockDim.x) A[i][c] /= piv;
        __syncthreads();
        for (int r = tid; r < IDIMn; r += blockDim.x) {
            if (r != i) {
                float fct = A[r][i];
                for (int c = 0; c < 2 * IDIMn; ++c) A[r][c] -= fct * A[i][c];
            }
        }
        __syncthreads();
    }
    for (int e = tid; e < IDIMn * IDIMn; e += blockDim.x)
        invW[(size_t)f * IDIMn * IDIMn + e] = A[e / IDIMn][IDIMn + e % IDIMn];
    if (tid == 0) logdet[f] = ld;
}

__global__ void ldj_const_kernel(float* __restrict__ ldj,
                                 const float* __restrict__ logdet,
                                 const float* __restrict__ logs) {
    if (threadIdx.x == 0 && blockIdx.x == 0) {
        float s = logdet[0];
        for (int c = 0; c < IDIMn; ++c) s += logs[c];
        float add = (float)Nptsn * s;
        for (int b = 0; b < Bn; ++b) ldj[b] += add;
    }
}

// ---------------------------------------------------------------------------
// gauss_logp and final NLL
// ---------------------------------------------------------------------------
__global__ void gauss_logp_kernel(const float* __restrict__ Z, int C,
                                  float* __restrict__ out) {
    int b = blockIdx.x;
    __shared__ float red[256];
    float s = 0.0f;
    for (int t = threadIdx.x; t < Nptsn * C; t += blockDim.x) {
        float v = Z[(size_t)b * Nptsn * C + t];
        s += v * v + LOG2PIc;
    }
    red[threadIdx.x] = s;
    __syncthreads();
    for (int off = 128; off > 0; off >>= 1) {
        if ((int)threadIdx.x < off) red[threadIdx.x] += red[threadIdx.x + off];
        __syncthreads();
    }
    if (threadIdx.x == 0) out[b] = -0.5f * red[0];
}

__global__ void nll_kernel(const float* __restrict__ lph,
                           const float* __restrict__ ldj,
                           const float* __restrict__ lpe,
                           const float* __restrict__ stepldj,
                           float* __restrict__ out) {
    if (threadIdx.x == 0 && blockIdx.x == 0) {
        float s = 0.0f;
        for (int b = 0; b < Bn; ++b) {
            float aug_ldj = lpe[b] - stepldj[b];
            s += lph[b] + ldj[b] - aug_ldj;
        }
        out[0] = -s / (float)Bn;
    }
}

// ===========================================================================
// Host orchestration
// ===========================================================================
namespace {

struct NetP {
    const float *w1, *b1, *g1, *be1, *w2, *b2, *g2, *be2, *w3, *b3;
    int lin;
};
struct FlowP {
    const float *W, *bias, *logs;
    NetP c1s, c1t, c2s, c2t;
    int knn;
};
struct AugP { const float *b1, *bs, *bt, *w1, *ws, *wt; };

struct WS {
    float *bufA, *bufB, *x1buf, *featb, *ybuf, *cat74;
    float *h64, *h64b, *sbuf, *tbuf, *mean, *var;
    float *Wpack, *invW, *logdet, *ldj, *stepldj, *lpe, *lph;
    float *bigA, *bigB;
    int   *idxAll;
    _Float16 *Ap, *BpG, *Bp1, *Bp2, *Bp3;
};

inline int cdiv(long a, long b) { return (int)((a + b - 1) / b); }

inline void launch_pack_a(const float* X, int M, int K, int Kt, _Float16* Ap,
                          hipStream_t st) {
    long total = (long)(M >> 4) * Kt * 32;
    pack_a_kernel<<<cdiv(total, 256), 256, 0, st>>>(X, M, K, Kt, Ap);
}
inline void launch_pack_b(const float* W, int K, int N, int Kt, _Float16* Bp,
                          hipStream_t st) {
    long total = (long)cdiv(N, 16) * Kt * 32;
    pack_b_kernel<<<cdiv(total, 256), 256, 0, st>>>(W, K, N, Kt, Bp);
}
inline void launch_gemm_packed(const _Float16* Ap, const _Float16* Bp,
                               const float* bias, float* Y,
                               int M, int N, int Kt, int relu, hipStream_t st) {
    long waves = (long)(M >> 4) * cdiv(N, 16);
    gemm_wmma_kernel<<<cdiv(waves * 32, 256), 256, 0, st>>>(Ap, Bp, bias, Y, M, N, Kt, relu);
}
// Full layer: pack A + pack B + GEMM (for non-hot paths).
inline void run_layer(const float* X, const float* W, const float* bias, float* Y,
                      int M, int K, int N, int relu, const WS& w, hipStream_t st) {
    int Kt = cdiv(K, 32);
    launch_pack_a(X, M, K, Kt, w.Ap, st);
    launch_pack_b(W, K, N, Kt, w.BpG, st);
    launch_gemm_packed(w.Ap, w.BpG, bias, Y, M, N, Kt, relu, st);
}

void eval_net_knn(const NetP& n, const WS& w, const int* idx, int k,
                  float* outC2, hipStream_t st) {
    const int Kt1 = cdiv(3 * C1n, 32);   // 36 -> 2
    const int Kt2 = cdiv(HIDn, 32);      // 64 -> 2
    launch_pack_b(n.w1, 3 * C1n, HIDn, Kt1, w.Bp1, st);
    launch_pack_b(n.w2, HIDn,    HIDn, Kt2, w.Bp2, st);
    launch_pack_b(n.w3, HIDn,    C2n,  Kt2, w.Bp3, st);
    for (int p0 = 0; p0 < Pn; p0 += CHUNKP) {
        int np = CHUNKP;
        int rows = np * k;               // multiple of 16
        knn_build_kernel<<<cdiv((long)rows, 256), 256, 0, st>>>(w.x1buf, idx, k, p0, np, w.bigA);
        launch_pack_a(w.bigA, rows, 3 * C1n, Kt1, w.Ap, st);
        launch_gemm_packed(w.Ap, w.Bp1, n.b1, w.bigB, rows, HIDn, Kt1, 1, st);
        launch_pack_a(w.bigB, rows, HIDn, Kt2, w.Ap, st);
        launch_gemm_packed(w.Ap, w.Bp2, n.b2, w.bigA, rows, HIDn, Kt2, 1, st);
        launch_pack_a(w.bigA, rows, HIDn, Kt2, w.Ap, st);
        launch_gemm_packed(w.Ap, w.Bp3, n.b3, w.bigB, rows, C2n, Kt2, 0, st);
        rowmax_kernel<<<cdiv((long)np * C2n, 256), 256, 0, st>>>(w.bigB, k, C2n, p0, np, outC2);
    }
}

void eval_net_lin(const NetP& n, const WS& w, float* outC2, hipStream_t st) {
    run_layer(w.x1buf, n.w1, n.b1, w.h64, Pn, C1n, HIDn, 0, w, st);
    bn_stats_kernel<<<HIDn, 256, 0, st>>>(w.h64, HIDn, w.mean, w.var);
    bn_apply_relu_kernel<<<cdiv((long)Pn * HIDn, 256), 256, 0, st>>>(w.h64, HIDn, w.mean, w.var, n.g1, n.be1);
    run_layer(w.h64, n.w2, n.b2, w.h64b, Pn, HIDn, HIDn, 0, w, st);
    bn_stats_kernel<<<HIDn, 256, 0, st>>>(w.h64b, HIDn, w.mean, w.var);
    bn_apply_relu_kernel<<<cdiv((long)Pn * HIDn, 256), 256, 0, st>>>(w.h64b, HIDn, w.mean, w.var, n.g2, n.be2);
    run_layer(w.h64b, n.w3, n.b3, outC2, Pn, HIDn, C2n, 0, w, st);
}

void coupling(const NetP& s, const NetP& t, float* state, const int* idx, int k,
              bool fwd, const WS& w, hipStream_t st) {
    slice12_kernel<<<cdiv((long)Pn * C1n, 256), 256, 0, st>>>(state, w.x1buf);
    if (idx) {
        eval_net_knn(s, w, idx, k, w.sbuf, st);
        eval_net_knn(t, w, idx, k, w.tbuf, st);
    } else {
        eval_net_lin(s, w, w.sbuf, st);
        eval_net_lin(t, w, w.tbuf, st);
    }
    int blocks = cdiv((long)Pn * C2n, 256);
    if (fwd) couple_fwd_kernel<<<blocks, 256, 0, st>>>(state, w.sbuf, w.tbuf, w.ldj);
    else     couple_inv_kernel<<<blocks, 256, 0, st>>>(state, w.sbuf, w.tbuf);
}

} // namespace

extern "C" void kernel_launch(void* const* d_in, const int* in_sizes, int n_in,
                              void* d_out, int out_size, void* d_ws, size_t ws_size,
                              hipStream_t stream) {
    (void)in_sizes; (void)out_size; (void)ws_size;
    hipStream_t st = stream;

    // ---- unpack parameters (JAX tree-flatten order: dict keys sorted) ------
    auto F = [&](int i) -> const float* {
        return (const float*)d_in[(i < n_in) ? i : (n_in - 1)];
    };
    int ix = 0;
    const float* x   = F(ix++);   // (B,N,3)
    const float* eps = F(ix++);   // (B,N,20)

    AugP aug[3];
    for (int i = 0; i < 3; ++i) {  // keys sorted: b1,bs,bt,w1,ws,wt
        aug[i].b1 = F(ix++); aug[i].bs = F(ix++); aug[i].bt = F(ix++);
        aug[i].w1 = F(ix++); aug[i].ws = F(ix++); aug[i].wt = F(ix++);
    }
    auto read_net = [&](int lin) -> NetP {
        NetP n = {};
        n.lin = lin;
        // knn net sorted: b1,b2,b3,w1,w2,w3
        // lin net sorted: b1,b2,b3,be1,be2,g1,g2,w1,w2,w3
        n.b1 = F(ix++); n.b2 = F(ix++); n.b3 = F(ix++);
        if (lin) { n.be1 = F(ix++); n.be2 = F(ix++); n.g1 = F(ix++); n.g2 = F(ix++); }
        n.w1 = F(ix++); n.w2 = F(ix++); n.w3 = F(ix++);
        return n;
    };
    FlowP flows[NFLOWn];
    for (int i = 0; i < NFLOWn; ++i) {  // keys sorted: W,bias,c1(s,t),c2(s,t),logs
        int knn = (i < 3) || (i % 3 == 0);
        flows[i].knn  = knn;
        flows[i].W    = F(ix++);
        flows[i].bias = F(ix++);
        flows[i].c1s  = read_net(!knn);
        flows[i].c1t  = read_net(!knn);
        flows[i].c2s  = read_net(!knn);
        flows[i].c2t  = read_net(!knn);
        flows[i].logs = F(ix++);
    }
    NetP shallow = read_net(0);   // b1,b2,b3,w1,w2,w3

    // ---- workspace layout (aligned to 64B) ---------------------------------
    float* wp = (float*)d_ws;
    size_t off = 0;
    auto alloc = [&](size_t nfl) {
        float* p = wp + off;
        off += (nfl + 15) & ~(size_t)15;   // keep 64B alignment
        return p;
    };
    WS w;
    w.bufA   = alloc((size_t)Pn * IDIMn);
    w.bufB   = alloc((size_t)Pn * IDIMn);
    w.x1buf  = alloc((size_t)Pn * C1n);
    w.featb  = alloc((size_t)Pn * HIDn);
    w.ybuf   = alloc((size_t)Pn * AUG_CHn);
    w.cat74  = alloc((size_t)Pn * (AUGHn + HIDn));
    w.h64    = alloc((size_t)Pn * HIDn);
    w.h64b   = alloc((size_t)Pn * HIDn);
    w.sbuf   = alloc((size_t)Pn * C2n);
    w.tbuf   = alloc((size_t)Pn * C2n);
    w.mean   = alloc(HIDn);
    w.var    = alloc(HIDn);
    w.Wpack  = alloc((size_t)NFLOWn * IDIMn * IDIMn);
    w.invW   = alloc((size_t)NFLOWn * IDIMn * IDIMn);
    w.logdet = alloc(NFLOWn);
    w.ldj    = alloc(Bn);
    w.stepldj= alloc(Bn);
    w.lpe    = alloc(Bn);
    w.lph    = alloc(Bn);
    w.idxAll = (int*)alloc((size_t)Pn * (8 + 16 + 24 + 16 + 16 + 16));
    w.bigA   = alloc((size_t)CHUNKP * KMAXn * HIDn);
    w.bigB   = alloc((size_t)CHUNKP * KMAXn * HIDn);
    // f16 packed-operand buffers (half = 2B -> nfl/2 floats, padded)
    w.Ap  = (_Float16*)alloc(((size_t)CHUNKP * KMAXn * 96) / 2 + 16);
    w.BpG = (_Float16*)alloc(8192 / 2);
    w.Bp1 = (_Float16*)alloc(8192 / 2);
    w.Bp2 = (_Float16*)alloc(8192 / 2);
    w.Bp3 = (_Float16*)alloc(8192 / 2);

    // per-flow KNN idx slots: flows 0,1,2 (k=8,16,24 on x), 3,6,9 (k=16 on h)
    int* idxSlot[NFLOWn] = {};
    int  kSlot[NFLOWn] = {};
    {
        size_t io = 0;
        int ks[6] = {8, 16, 24, 16, 16, 16};
        int fl[6] = {0, 1, 2, 3, 6, 9};
        for (int s = 0; s < 6; ++s) {
            idxSlot[fl[s]] = w.idxAll + io;
            kSlot[fl[s]]   = ks[s];
            io += (size_t)Pn * ks[s];
        }
    }

    // ---- init accumulators -------------------------------------------------
    hipMemsetAsync(w.ldj,     0, Bn * sizeof(float), st);
    hipMemsetAsync(w.stepldj, 0, Bn * sizeof(float), st);

    // ---- mixing matrix pack + invert + logdet ------------------------------
    for (int i = 0; i < NFLOWn; ++i)
        hipMemcpyAsync(w.Wpack + (size_t)i * IDIMn * IDIMn, flows[i].W,
                       IDIMn * IDIMn * sizeof(float), hipMemcpyDeviceToDevice, st);
    invert_kernel<<<NFLOWn, 64, 0, st>>>(w.Wpack, w.invW, w.logdet);

    // ---- shallow feature MLP ----------------------------------------------
    run_layer(x, shallow.w1, shallow.b1, w.h64,  Pn, IN_CHn, 32, 1, w, st);
    run_layer(w.h64, shallow.w2, shallow.b2, w.h64b, Pn, 32, 32, 1, w, st);
    run_layer(w.h64b, shallow.w3, shallow.b3, w.featb, Pn, 32, HIDn, 0, w, st);

    // ---- augmentation steps ------------------------------------------------
    hipMemcpyAsync(w.ybuf, eps, (size_t)Pn * AUG_CHn * sizeof(float),
                   hipMemcpyDeviceToDevice, st);
    for (int i = 0; i < 3; ++i) {
        int rev = (i == 1);
        aug_concat_kernel<<<cdiv((long)Pn * (AUGHn + HIDn), 256), 256, 0, st>>>(
            w.ybuf, w.featb, w.cat74, rev);
        run_layer(w.cat74, aug[i].w1, aug[i].b1, w.h64, Pn, AUGHn + HIDn, HIDn, 1, w, st);
        run_layer(w.h64, aug[i].ws, aug[i].bs, w.sbuf, Pn, HIDn, AUGHn, 0, w, st);
        run_layer(w.h64, aug[i].wt, aug[i].bt, w.tbuf, Pn, HIDn, AUGHn, 0, w, st);
        aug_combine_kernel<<<cdiv((long)Pn * AUGHn, 256), 256, 0, st>>>(
            w.ybuf, w.sbuf, w.tbuf, rev, w.stepldj);
    }
    gauss_logp_kernel<<<Bn, 256, 0, st>>>(eps, AUG_CHn, w.lpe);

    // ---- h = concat(x, y) --------------------------------------------------
    concat_xy_kernel<<<cdiv((long)Pn * IDIMn, 256), 256, 0, st>>>(x, w.ybuf, w.bufA);
    float* cur = w.bufA;
    float* alt = w.bufB;

    // KNN on original xyz for flows 0..2
    for (int i = 0; i < 3; ++i)
        knn_kernel<<<cdiv((long)Pn, 128), 128, 0, st>>>(x, IN_CHn, kSlot[i], idxSlot[i]);

    // ---- forward flows -----------------------------------------------------
    for (int i = 0; i < NFLOWn; ++i) {
        if (i >= 3 && i % 3 == 0)  // flows 3,6,9: KNN on current h
            knn_kernel<<<cdiv((long)Pn, 128), 128, 0, st>>>(cur, IDIMn, kSlot[i], idxSlot[i]);
        run_layer(cur, flows[i].W, nullptr, alt, Pn, IDIMn, IDIMn, 0, w, st);
        affine_fwd_kernel<<<cdiv((long)Pn * IDIMn, 256), 256, 0, st>>>(alt, flows[i].logs, flows[i].bias);
        ldj_const_kernel<<<1, 32, 0, st>>>(w.ldj, w.logdet + i, flows[i].logs);
        coupling(flows[i].c1s, flows[i].c1t, alt, idxSlot[i], kSlot[i], true, w, st);
        reverse_kernel<<<cdiv((long)Pn * IDIMn, 256), 256, 0, st>>>(alt, cur);
        coupling(flows[i].c2s, flows[i].c2t, cur, idxSlot[i], kSlot[i], true, w, st);
        // cur holds flow output; alt is free for next iteration
    }

    // ---- log-likelihood ----------------------------------------------------
    gauss_logp_kernel<<<Bn, 256, 0, st>>>(cur, IDIMn, w.lph);
    nll_kernel<<<1, 32, 0, st>>>(w.lph, w.ldj, w.lpe, w.stepldj,
                                 (float*)d_out + (size_t)Pn * IN_CHn);

    // ---- inverse pass: z = h with last CUT channels zeroed -----------------
    zero_tail_kernel<<<cdiv((long)Pn * IN_CHn, 256), 256, 0, st>>>(cur);
    for (int i = NFLOWn - 1; i >= 0; --i) {
        coupling(flows[i].c2s, flows[i].c2t, cur, idxSlot[i], kSlot[i], false, w, st);
        reverse_kernel<<<cdiv((long)Pn * IDIMn, 256), 256, 0, st>>>(cur, alt);
        { float* tmp = cur; cur = alt; alt = tmp; }
        coupling(flows[i].c1s, flows[i].c1t, cur, idxSlot[i], kSlot[i], false, w, st);
        affine_inv_kernel<<<cdiv((long)Pn * IDIMn, 256), 256, 0, st>>>(cur, flows[i].logs, flows[i].bias);
        run_layer(cur, w.invW + (size_t)i * IDIMn * IDIMn, nullptr, alt, Pn, IDIMn, IDIMn, 0, w, st);
        { float* tmp = cur; cur = alt; alt = tmp; }
    }

    // ---- write z[..., :3] --------------------------------------------------
    out_kernel<<<cdiv((long)Pn * IN_CHn, 256), 256, 0, st>>>(cur, (float*)d_out);
}
